// MergeHeads_60447369724369
// MI455X (gfx1250) — matrix operand order repaired and checked by reading, not compiled, not run on hardware
//
#include <hip/hip_runtime.h>

typedef _Float16 v8h  __attribute__((ext_vector_type(8)));
typedef _Float16 v16h __attribute__((ext_vector_type(16)));
typedef float    v8f  __attribute__((ext_vector_type(8)));

namespace {
constexpr int kDModel = 2048;
constexpr int kDIn    = 2048;
constexpr int kS      = 2048;

constexpr int BM = 128;       // output rows per block
constexpr int BN = 128;       // output cols per block
constexpr int BK = 32;        // K per stage = one wmma k-step
constexpr int SAH = 40;       // halves per A row in LDS (32 + 8 pad; 20 words ≡ 4 mod 8 -> conflict-free b128)
constexpr int SWH = 40;       // halves per W column in LDS
constexpr int NST = kDIn / BK;  // 64
}

#define CAT16(lo8, hi8) \
    __builtin_shufflevector((lo8), (hi8), 0,1,2,3,4,5,6,7,8,9,10,11,12,13,14,15)

__device__ __forceinline__ void split1(float x, _Float16& h, _Float16& l) {
    h = (_Float16)x;                 // top ~11 mantissa bits
    l = (_Float16)(x - (float)h);    // next ~11 bits
}
__device__ __forceinline__ void split4(const float4& v, _Float16* h, _Float16* l) {
    split1(v.x, h[0], l[0]); split1(v.y, h[1], l[1]);
    split1(v.z, h[2], l[2]); split1(v.w, h[3], l[3]);
}

__global__ __launch_bounds__(256)
void banked_vlinear_wmma_f16x2(const float* __restrict__ x,        // (B, S, D_IN) [head merge = no-op]
                               const int*   __restrict__ sel_idx,  // (B, 2)
                               const float* __restrict__ sel_prob, // (B, 2)
                               const float* __restrict__ weight,   // (16, D_IN, D_MODEL)
                               const float* __restrict__ bias,     // (16, D_MODEL)
                               float*       __restrict__ out)      // (B, S, D_MODEL)
{
    // A stored [m][k] (row-major), W stored [n][k] (column-major) -- both hi and lo planes.
    __shared__ __align__(16) _Float16 sAh[BM * SAH];
    __shared__ __align__(16) _Float16 sAl[BM * SAH];
    __shared__ __align__(16) _Float16 sWh[BN * SWH];
    __shared__ __align__(16) _Float16 sWl[BN * SWH];

    const int tid  = threadIdx.x;
    const int lane = tid & 31;
    const int wave = tid >> 5;
    const int wm   = wave >> 2;     // 0..1 -> 64-row slab
    const int wn   = wave & 3;      // 0..3 -> 32-col slab
    const int l16  = lane & 15;
    const int hi   = lane >> 4;     // half-wave selector

    const int bn0 = blockIdx.x * BN;
    const int bm0 = blockIdx.y * BM;
    const int b   = blockIdx.z;

    const int   i0 = sel_idx[b * 2 + 0];
    const int   i1 = sel_idx[b * 2 + 1];
    const float p0 = sel_prob[b * 2 + 0];
    const float p1 = sel_prob[b * 2 + 1];

    const float* xb  = x + (long)b * kS * kDIn;
    const float* wb0 = weight + (long)i0 * kDIn * kDModel;
    const float* wb1 = weight + (long)i1 * kDIn * kDModel;

    // cooperative staging coordinates
    const int arow = tid >> 1;           // A row 0..127
    const int akc  = (tid & 1) * 16;     // A k-chunk {0,16}
    const int wcol = tid & 127;          // W output column 0..127
    const int wkc  = (tid >> 7) * 16;    // W k-chunk {0,16}

    float4 av[4];     // 16 A elements (one row chunk)
    float  wv[16];    // 16 blended W elements (one column chunk)

    auto g_load = [&](int stage) {
        const long k0 = (long)stage * BK;
        const float* ap = xb + (long)(bm0 + arow) * kDIn + k0 + akc;
        av[0] = *(const float4*)(ap + 0);
        av[1] = *(const float4*)(ap + 4);
        av[2] = *(const float4*)(ap + 8);
        av[3] = *(const float4*)(ap + 12);
        // k-strided, lane-coalesced reads of both selected banks; fused superposition
        const float* wp0 = wb0 + (k0 + wkc) * kDModel + bn0 + wcol;
        const float* wp1 = wb1 + (k0 + wkc) * kDModel + bn0 + wcol;
        #pragma unroll
        for (int j = 0; j < 16; ++j)
            wv[j] = fmaf(p0, wp0[(long)j * kDModel], p1 * wp1[(long)j * kDModel]);
    };

    auto l_store = [&]() {
        alignas(16) _Float16 ah[16], al[16], wh[16], wl[16];
        #pragma unroll
        for (int j = 0; j < 4; ++j) split4(av[j], &ah[4 * j], &al[4 * j]);
        #pragma unroll
        for (int j = 0; j < 16; ++j) split1(wv[j], wh[j], wl[j]);
        *(v8h*)&sAh[arow * SAH + akc    ] = *(const v8h*)&ah[0];
        *(v8h*)&sAh[arow * SAH + akc + 8] = *(const v8h*)&ah[8];
        *(v8h*)&sAl[arow * SAH + akc    ] = *(const v8h*)&al[0];
        *(v8h*)&sAl[arow * SAH + akc + 8] = *(const v8h*)&al[8];
        *(v8h*)&sWh[wcol * SWH + wkc    ] = *(const v8h*)&wh[0];
        *(v8h*)&sWh[wcol * SWH + wkc + 8] = *(const v8h*)&wh[8];
        *(v8h*)&sWl[wcol * SWH + wkc    ] = *(const v8h*)&wl[0];
        *(v8h*)&sWl[wcol * SWH + wkc + 8] = *(const v8h*)&wl[8];
    };

    v8f acc[4][2];
    #pragma unroll
    for (int mi = 0; mi < 4; ++mi)
        #pragma unroll
        for (int ni = 0; ni < 2; ++ni)
            acc[mi][ni] = v8f{0.f, 0.f, 0.f, 0.f, 0.f, 0.f, 0.f, 0.f};

    g_load(0);
    l_store();
    __syncthreads();

    for (int s = 0; s < NST; ++s) {
        if (s + 1 < NST) g_load(s + 1);   // register-prefetch next stage during compute

        // ---- B fragments: lane<16 holds K 0..15, lane>=16 holds K 16..31 of column l16 ----
        v16h bh[2], bl[2];
        #pragma unroll
        for (int ni = 0; ni < 2; ++ni) {
            const int base = (wn * 32 + ni * 16 + l16) * SWH + hi * 16;
            v8h h0 = *(const v8h*)&sWh[base];
            v8h h1 = *(const v8h*)&sWh[base + 8];
            v8h q0 = *(const v8h*)&sWl[base];
            v8h q1 = *(const v8h*)&sWl[base + 8];
            bh[ni] = CAT16(h0, h1);
            bl[ni] = CAT16(q0, q1);
        }

        // ---- A fragments + 3 WMMAs per (mi,ni): Xh*Wh + Xh*Wl + Xl*Wh ----
        #pragma unroll
        for (int mi = 0; mi < 4; ++mi) {
            const int rbase = (wm * 64 + mi * 16 + l16) * SAH;
            // lane<16: elements 0..15 <-> K0..15 ; lane>=16: K16..31
            v8h a0 = *(const v8h*)&sAh[rbase + hi * 16];
            v8h a1 = *(const v8h*)&sAh[rbase + hi * 16 + 8];
            v8h c0 = *(const v8h*)&sAl[rbase + hi * 16];
            v8h c1 = *(const v8h*)&sAl[rbase + hi * 16 + 8];
            v16h ah = CAT16(a0, a1);
            v16h al = CAT16(c0, c1);
            #pragma unroll
            for (int ni = 0; ni < 2; ++ni) {
                acc[mi][ni] = __builtin_amdgcn_wmma_f32_16x16x32_f16(
                    false, ah, false, bh[ni], (short)0, acc[mi][ni], false, false);
                acc[mi][ni] = __builtin_amdgcn_wmma_f32_16x16x32_f16(
                    false, ah, false, bl[ni], (short)0, acc[mi][ni], false, false);
                acc[mi][ni] = __builtin_amdgcn_wmma_f32_16x16x32_f16(
                    false, al, false, bh[ni], (short)0, acc[mi][ni], false, false);
            }
        }

        __syncthreads();                 // everyone done reading the single LDS buffer
        if (s + 1 < NST) {
            l_store();                   // publish prefetched stage
            __syncthreads();
        }
    }

    // ---- epilogue: add blended bias; C/D layout: VGPR r -> rows r / r+8, N = lane ----
    #pragma unroll
    for (int ni = 0; ni < 2; ++ni) {
        const int col = bn0 + wn * 32 + ni * 16 + l16;
        const float bsel = fmaf(p0, bias[(long)i0 * kDModel + col],
                                p1 * bias[(long)i1 * kDModel + col]);
        #pragma unroll
        for (int mi = 0; mi < 4; ++mi) {
            const int rb = bm0 + wm * 64 + mi * 16 + hi * 8;
            #pragma unroll
            for (int r = 0; r < 8; ++r)
                out[((long)b * kS + rb + r) * kDModel + col] = acc[mi][ni][r] + bsel;
        }
    }
}

extern "C" void kernel_launch(void* const* d_in, const int* in_sizes, int n_in,
                              void* d_out, int out_size, void* d_ws, size_t ws_size,
                              hipStream_t stream) {
    (void)in_sizes; (void)n_in; (void)d_ws; (void)ws_size; (void)out_size;
    const float* x        = (const float*)d_in[0];
    const int*   sel_idx  = (const int*)  d_in[1];
    const float* sel_prob = (const float*)d_in[2];
    const float* weight   = (const float*)d_in[3];
    const float* bias     = (const float*)d_in[4];
    float*       out      = (float*)d_out;

    dim3 grid(kDModel / BN, kS / BM, 2 /*B*/);
    dim3 block(256);
    banked_vlinear_wmma_f16x2<<<grid, block, 0, stream>>>(x, sel_idx, sel_prob, weight, bias, out);
}
